// LSTM_Decoder_35811437314186
// MI455X (gfx1250) — matrix operand order repaired and checked by reading, not compiled, not run on hardware
//
#include <hip/hip_runtime.h>
#include <hip/hip_fp16.h>

// ---------------------------------------------------------------------------
// LSTM decoder, restructured for MI455X (gfx1250):
//   K0: gather embeddings -> f16, pad mask; convert W_ih / W_fc -> f16; init state
//   K1: G[T,4H] = X[T,E] @ W_ih^T + (b_ih + b_hh)          (WMMA f16->f32)
//   K2: persistent recurrence: h @ W_hh^T (W_hh slice LDS-resident via
//       global_load_async_to_lds_b128), LSTM cell, pad-masked state carry.
//   K3: Out[T,V] = Hs[T,H] @ W_fc^T + b_fc                 (WMMA f16->f32, NT stores)
// ---------------------------------------------------------------------------

typedef __attribute__((ext_vector_type(16))) _Float16 v16h;
typedef __attribute__((ext_vector_type(8)))  _Float16 v8h;
typedef __attribute__((ext_vector_type(4)))  _Float16 v4h;
typedef __attribute__((ext_vector_type(8)))  float    v8f;

constexpr int Vv = 32000;
constexpr int Ee = 512;
constexpr int Hh = 1024;
constexpr int Tt = 512;
constexpr int G4 = 4096;          // 4*H

constexpr int NWG      = 64;      // workgroups in recurrence kernel
constexpr int RTHREADS = 256;
constexpr int LDSW     = Hh + 4;  // padded LDS row stride (16B-aligned, bank shift)

// Async memory->LDS copy of 16 bytes per lane (gfx1250 GLOBAL_LOAD_ASYNC_TO_LDS_B128,
// tracked by ASYNCcnt). ldsOff is the wave-relative LDS byte address: the low 32 bits
// of a generic pointer into LDS (ISA 10.2, LDS aperture).
__device__ __forceinline__ void async_copy_b128(unsigned ldsOff, const void* gaddr) {
    asm volatile("global_load_async_to_lds_b128 %0, %1, off"
                 :: "v"(ldsOff), "v"(gaddr) : "memory");
}
__device__ __forceinline__ void async_wait0() {
    asm volatile("s_wait_asynccnt 0" ::: "memory");
}

// ---------------------------------------------------------------------------
// K0 helpers
// ---------------------------------------------------------------------------
__global__ void gather_embed_kernel(const float* __restrict__ emb,
                                    const int* __restrict__ inputs,
                                    _Float16* __restrict__ X16,
                                    int* __restrict__ pad)
{
    int idx = blockIdx.x * blockDim.x + threadIdx.x;       // over T*E
    if (idx < Tt * Ee) {
        int t = idx >> 9;                                  // E = 512
        int e = idx & (Ee - 1);
        X16[idx] = (_Float16)emb[(size_t)inputs[t] * Ee + e];
    }
    if (idx < Tt) pad[idx] = (inputs[idx] == 0) ? 1 : 0;
}

__global__ void cvt_f32_f16_v4_kernel(const float* __restrict__ src,
                                      _Float16* __restrict__ dst, int n4)
{
    int i = blockIdx.x * blockDim.x + threadIdx.x;
    if (i < n4) {
        float4 v = ((const float4*)src)[i];
        v4h h;
        h[0] = (_Float16)v.x; h[1] = (_Float16)v.y;
        h[2] = (_Float16)v.z; h[3] = (_Float16)v.w;
        ((v4h*)dst)[i] = h;
    }
}

__global__ void init_state_kernel(const float* __restrict__ hidden,
                                  float* __restrict__ h_buf,
                                  unsigned* __restrict__ bar)
{
    int i = blockIdx.x * blockDim.x + threadIdx.x;
    if (i < Hh) h_buf[i] = hidden[i];                      // parity-0 buffer
    if (i == 0) *bar = 0u;
}

// ---------------------------------------------------------------------------
// WMMA GEMM: C[M,N] = A[M,K](f16,row-major) * B[N,K](f16,row-major)^T
//            + bias0[N] (+ bias1[N]); fp32 accumulate, nontemporal C stores.
// One wave -> 64x64 tile (4 A-frags x 4 B-frags, 16 accumulators) for 4x reuse
// of BOTH operands; block = 4 waves -> 64x256 tile.
// __launch_bounds__(128, 1): ~212 live VGPRs (128 accum + 64 frags + addr) --
// allow the full register file so accumulators are NOT spilled to scratch.
// Fragment packing used here (wave32, contiguous k, same order for A and B):
//   A 16x32 f16: lane<16 holds K 0..15, lane>=16 holds K 16..31
//   B 32x16 f16: lane = column; lane<16 holds K 0..15, lane>=16 holds K 16..31
//   C 16x16 f32: VGPR d -> M = d + (lane>=16 ? 8 : 0), N = lane&15
// ---------------------------------------------------------------------------
__global__ void __launch_bounds__(128, 1)
wmma_gemm_f16f32_kernel(const _Float16* __restrict__ A,
                        const _Float16* __restrict__ B,
                        const float* __restrict__ bias0,
                        const float* __restrict__ bias1,
                        float* __restrict__ C,
                        int M, int N, int K)
{
    const int lane = threadIdx.x & 31;
    const int wv   = threadIdx.x >> 5;
    const int m0   = blockIdx.y << 6;                  // 64 rows per block
    const int n0   = (blockIdx.x << 8) + (wv << 6);    // 64 cols per wave

    const int row = lane & 15;
    const int ak  = (lane >> 4) << 4;     // 0 or 16
    const int bk  = (lane >> 4) << 4;     // 0 or 16

    const _Float16* pa = A + (size_t)(m0 + row) * K + ak;
    const _Float16* pb = B + (size_t)(n0 + row) * K + bk;
    const size_t fstep = (size_t)16 * K;  // 16 rows/cols

    v8f acc[4][4] = {};                   // [mi][ni]

    for (int k0 = 0; k0 < K; k0 += 32) {
        v16h a[4], b[4];
        #pragma unroll
        for (int i = 0; i < 4; ++i) {
            a[i] = *(const v16h*)(pa + i * fstep + k0);
            b[i] = *(const v16h*)(pb + i * fstep + k0);
        }
        __builtin_prefetch((const void*)(pb + k0 + 256), 0, 1);  // global_prefetch_b8
        #pragma unroll
        for (int mi = 0; mi < 4; ++mi)
            #pragma unroll
            for (int ni = 0; ni < 4; ++ni)
                acc[mi][ni] = __builtin_amdgcn_wmma_f32_16x16x32_f16(
                    false, a[mi], false, b[ni], (short)0, acc[mi][ni], false, false);
    }

    const int crow = (lane >> 4) << 3;
    const int ccol = lane & 15;
    #pragma unroll
    for (int ni = 0; ni < 4; ++ni) {
        int col  = n0 + (ni << 4) + ccol;
        float bs = bias0[col] + (bias1 ? bias1[col] : 0.0f);
        #pragma unroll
        for (int mi = 0; mi < 4; ++mi) {
            float* cp = C + (size_t)(m0 + (mi << 4) + crow) * N + col;
            #pragma unroll
            for (int d = 0; d < 8; ++d)
                __builtin_nontemporal_store(acc[mi][ni][d] + bs, cp + (size_t)d * N);
        }
    }
}

// ---------------------------------------------------------------------------
// K2: persistent recurrence. 64 WGs; WG b owns hidden units [16b,16b+16) and
// the 64 matching W_hh rows (i/f/g/o), staged into LDS (~257 KB fp32) with
// async-to-LDS B128 copies and held resident for all T steps. Double-buffered
// global h vector; monotonic atomic grid barrier (+cluster-barrier fast path).
// ---------------------------------------------------------------------------
__device__ __forceinline__ float sigmoidf_(float x) {
    return 1.0f / (1.0f + __expf(-x));
}

__global__ void lstm_recurrence_kernel(const float* __restrict__ W_hh,
                                       const float* __restrict__ G,      // [T,4H] incl. both biases
                                       const float* __restrict__ cell,
                                       const int* __restrict__ pad,
                                       float* __restrict__ h_buf,        // [2][H]
                                       _Float16* __restrict__ Hs16,      // [T][H]
                                       unsigned* __restrict__ bar)
{
    extern __shared__ float smem[];
    float* Wsl = smem;                   // 64 * LDSW
    float* hsh = Wsl + 64 * LDSW;        // H
    float* red = hsh + Hh;               // RTHREADS
    float* gts = red + RTHREADS;         // 64
    float* cs  = gts + 64;               // 16

    const int tid = threadIdx.x;
    const int bj  = blockIdx.x * 16;     // hidden-slice base

    // Stage 64 W_hh rows (local row r = gate*16 + j) directly into LDS with
    // async B128 copies: 16384 16-byte chunks, 16 B per lane per op.
    for (int c = tid; c < 64 * (Hh / 4); c += RTHREADS) {
        int r = c >> 8;                  // 256 chunks per row
        int q = c & 255;
        int g = r >> 4, j = r & 15;
        const float* gp = W_hh + (size_t)(g * Hh + bj + j) * Hh + q * 4;
        unsigned loff = (unsigned)(uintptr_t)(const void*)&Wsl[r * LDSW + q * 4];
        async_copy_b128(loff, gp);
    }
    async_wait0();
    if (tid < 16) cs[tid] = cell[bj + tid];
    __syncthreads();

    const int r   = tid >> 2;            // 0..63: local gate-row
    const int sub = tid & 3;             // K quarter

    for (int t = 0; t < Tt; ++t) {
        // h vector (4 KB): one async B128 per lane, memory -> LDS
        {
            const float* hsrc = h_buf + (size_t)(t & 1) * Hh + tid * 4;
            unsigned loff = (unsigned)(uintptr_t)(const void*)&hsh[tid * 4];
            async_copy_b128(loff, hsrc);
            async_wait0();
        }
        __syncthreads();

        const float* wr = Wsl + r * LDSW + sub * 256;
        const float* hv = hsh + sub * 256;
        float s = 0.0f;
        #pragma unroll 8
        for (int k = 0; k < 256; ++k) s = fmaf(wr[k], hv[k], s);
        red[tid] = s;
        __syncthreads();

        if (tid < 64) {
            float tot = red[4 * tid] + red[4 * tid + 1] +
                        red[4 * tid + 2] + red[4 * tid + 3];
            int g = tid >> 4, j = tid & 15;
            gts[tid] = tot + G[(size_t)t * G4 + g * Hh + bj + j];
        }
        __syncthreads();

        if (tid < 16) {
            int j = tid;
            float ig = sigmoidf_(gts[j]);
            float fg = sigmoidf_(gts[16 + j]);
            float gg = tanhf(gts[32 + j]);
            float og = sigmoidf_(gts[48 + j]);
            float c  = cs[j];
            float cn = fg * c + ig * gg;
            float hn = og * tanhf(cn);
            Hs16[(size_t)t * Hh + bj + j] = (_Float16)hn;   // logits always use h_new
            int isPad   = pad[t];
            float hprev = hsh[bj + j];
            cs[j] = isPad ? c : cn;                         // state freezes on pad
            h_buf[(size_t)((t + 1) & 1) * Hh + bj + j] = isPad ? hprev : hn;
        }
        __threadfence();
        __syncthreads();

        if (tid == 0) {
            __builtin_amdgcn_s_cluster_barrier();           // NOP when not cluster-dispatched
            atomicAdd(bar, 1u);
            unsigned target = (unsigned)(t + 1) * (unsigned)gridDim.x;
            while (__hip_atomic_load(bar, __ATOMIC_ACQUIRE,
                                     __HIP_MEMORY_SCOPE_AGENT) < target)
                __builtin_amdgcn_s_sleep(1);
        }
        __syncthreads();
    }
}

// ---------------------------------------------------------------------------
// Host-side launch
// ---------------------------------------------------------------------------
extern "C" void kernel_launch(void* const* d_in, const int* in_sizes, int n_in,
                              void* d_out, int out_size, void* d_ws, size_t ws_size,
                              hipStream_t stream)
{
    const int*   inputs = (const int*)  d_in[0];
    const float* hidden = (const float*)d_in[1];
    const float* cell   = (const float*)d_in[2];
    // d_in[3] = generate_len (== T, compile-time constant here)
    const float* emb    = (const float*)d_in[4];
    const float* W_ih   = (const float*)d_in[5];
    const float* b_ih   = (const float*)d_in[6];
    const float* W_hh   = (const float*)d_in[7];
    const float* b_hh   = (const float*)d_in[8];
    const float* W_fc   = (const float*)d_in[9];
    const float* b_fc   = (const float*)d_in[10];
    float* out = (float*)d_out;

    char*  ws  = (char*)d_ws;
    size_t off = 0;
    auto alloc = [&](size_t bytes) -> void* {
        void* p = ws + off;
        off += bytes;
        off = (off + 255) & ~(size_t)255;
        return p;
    };

    _Float16* X16   = (_Float16*)alloc((size_t)Tt * Ee * 2);   // 0.5 MB
    _Float16* Wih16 = (_Float16*)alloc((size_t)G4 * Ee * 2);   // 4 MB
    _Float16* Wfc16 = (_Float16*)alloc((size_t)Vv * Hh * 2);   // 62.5 MB
    float*    Gbuf  = (float*)   alloc((size_t)Tt * G4 * 4);   // 8 MB
    _Float16* Hs16  = (_Float16*)alloc((size_t)Tt * Hh * 2);   // 1 MB
    float*    h_buf = (float*)   alloc((size_t)2 * Hh * 4);
    int*      pad   = (int*)     alloc((size_t)Tt * 4);
    unsigned* bar   = (unsigned*)alloc(256);

    // K0: init + gathers + f16 conversions
    init_state_kernel<<<(Hh + 255) / 256, 256, 0, stream>>>(hidden, h_buf, bar);
    gather_embed_kernel<<<(Tt * Ee + 255) / 256, 256, 0, stream>>>(emb, inputs, X16, pad);
    {
        int n4 = (G4 * Ee) / 4;
        cvt_f32_f16_v4_kernel<<<(n4 + 255) / 256, 256, 0, stream>>>(W_ih, Wih16, n4);
    }
    {
        int n4 = (Vv * Hh) / 4;
        cvt_f32_f16_v4_kernel<<<(n4 + 255) / 256, 256, 0, stream>>>(W_fc, Wfc16, n4);
    }

    // K1: G = X @ W_ih^T + (b_ih + b_hh)   [512 x 4096 x 512]
    {
        dim3 grid(G4 / 256, Tt / 64);
        wmma_gemm_f16f32_kernel<<<grid, 128, 0, stream>>>(
            X16, Wih16, b_ih, b_hh, Gbuf, Tt, G4, Ee);
    }

    // K2: sequential LSTM recurrence (persistent, LDS-resident W_hh slices)
    {
        size_t smem = (size_t)(64 * LDSW + Hh + RTHREADS + 64 + 16) * sizeof(float);
        lstm_recurrence_kernel<<<NWG, RTHREADS, smem, stream>>>(
            W_hh, Gbuf, cell, pad, h_buf, Hs16, bar);
    }

    // K3: Out = Hs @ W_fc^T + b_fc   [512 x 32000 x 1024]
    {
        dim3 grid(Vv / 256, Tt / 64);
        wmma_gemm_f16f32_kernel<<<grid, 128, 0, stream>>>(
            Hs16, Wfc16, b_fc, nullptr, out, Tt, Vv, Hh);
    }
}
